// IMPALA_49941879718457
// MI455X (gfx1250) — matrix operand order repaired and checked by reading, not compiled, not run on hardware
//
#include <hip/hip_runtime.h>
#include <hip/hip_bf16.h>
#include <math.h>

typedef __bf16 bf16;
typedef bf16  v8bf  __attribute__((ext_vector_type(8)));
typedef bf16  v16bf __attribute__((ext_vector_type(16)));
typedef float v8f   __attribute__((ext_vector_type(8)));

#define T_DIM 80
#define B_DIM 256
#define TBROWS 20480        // T*B
#define FEAT 3136
#define HID 256
#define ACT 18
#define PHID 128
#define XDIM 288            // 256 + 1 + 18 padded to multiple of 32
#define GATES 1024

// ---------------- workspace layout ----------------
// bf16 region (element offsets)
#define FCW_BF_OFF   0ull                         // 256*3136   = 802816
#define WIH_BF_OFF   802816ull                    // 1024*288   = 294912
#define WHH_BF_OFF   (WIH_BF_OFF + 294912ull)     // 1024*256   = 262144
#define PFCW_BF_OFF  (WHH_BF_OFF + 262144ull)     // 128*256    = 32768
#define XBF_OFF      (PFCW_BF_OFF + 32768ull)     // 20480*288  = 5898240
#define HSBF_OFF     (XBF_OFF + 5898240ull)       // 20480*256  = 5242880
#define BF_ELEMS     (HSBF_OFF + 5242880ull)
#define BF_BYTES     (BF_ELEMS * 2ull)            // 25,067,520 (multiple of 256)
// f32 region (element offsets from wsf)
#define BGATE_OFF    0ull                         // 1024
#define GX_OFF       1024ull                      // 20480*1024

// ---------------- helpers ----------------
__device__ __forceinline__ v8f vzero8f() {
    v8f z;
#pragma unroll
    for (int i = 0; i < 8; ++i) z[i] = 0.0f;
    return z;
}

__device__ __forceinline__ v16bf mkfrag(v8bf lo, v8bf hi) {
    v16bf a;
#pragma unroll
    for (int i = 0; i < 8; ++i) { a[i] = lo[i]; a[i + 8] = hi[i]; }
    return a;
}

__device__ __forceinline__ v8f wmma_bf16(v16bf a, v16bf b, v8f c) {
    // D(16x16 f32) = A(16x32 bf16) * B(32x16 bf16) + C
    return __builtin_amdgcn_wmma_f32_16x16x32_bf16(
        false, a, false, b, (short)0, c, false, false);
}

__device__ __forceinline__ float sigmoidf_(float x) {
    return 1.0f / (1.0f + __expf(-x));
}

// ---------------- K0: weight conversion / packing ----------------
__global__ void prep_kernel(const float* __restrict__ fc_w,
                            const float* __restrict__ w_ih,
                            const float* __restrict__ w_hh,
                            const float* __restrict__ pfc_w,
                            const float* __restrict__ b_ih,
                            const float* __restrict__ b_hh,
                            bf16* __restrict__ wsb,
                            float* __restrict__ wsf) {
    const int N0 = 802816, N1 = 294912, N2 = 262144, N3 = 32768, N4 = 1024;
    const int total = N0 + N1 + N2 + N3 + N4;
    for (int i = blockIdx.x * blockDim.x + threadIdx.x; i < total;
         i += gridDim.x * blockDim.x) {
        if (i < N0) {
            wsb[FCW_BF_OFF + i] = (bf16)fc_w[i];
        } else if (i < N0 + N1) {
            int j = i - N0;
            int n = j / XDIM, k = j % XDIM;
            wsb[WIH_BF_OFF + j] =
                (k < (HID + 1 + ACT)) ? (bf16)w_ih[n * (HID + 1 + ACT) + k] : (bf16)0.0f;
        } else if (i < N0 + N1 + N2) {
            int j = i - N0 - N1;
            wsb[WHH_BF_OFF + j] = (bf16)w_hh[j];
        } else if (i < N0 + N1 + N2 + N3) {
            int j = i - N0 - N1 - N2;
            wsb[PFCW_BF_OFF + j] = (bf16)pfc_w[j];
        } else {
            int j = i - N0 - N1 - N2 - N3;
            wsf[BGATE_OFF + j] = b_ih[j] + b_hh[j];
        }
    }
}

// ---------------- K1: fill x tail columns [256..287] ----------------
__global__ void fillx_kernel(const float* __restrict__ reward,
                             const float* __restrict__ action,
                             bf16* __restrict__ x_bf) {
    const int total = TBROWS * 32;
    for (int i = blockIdx.x * blockDim.x + threadIdx.x; i < total;
         i += gridDim.x * blockDim.x) {
        int tb = i >> 5, c = i & 31;
        float v = (c == 0) ? reward[tb]
                           : ((c <= ACT) ? action[(size_t)tb * ACT + (c - 1)] : 0.0f);
        x_bf[(size_t)tb * XDIM + HID + c] = (bf16)v;
    }
}

// ---------------- K2: FC GEMM  (20480 x 3136) @ (3136 x 256) ----------------
// block = 128(M) x 64(N), 8 waves; wave w owns M-strip [w*16, w*16+16) x 64 N
__global__ __launch_bounds__(256) void fc_gemm_kernel(
    const float* __restrict__ state, const float* __restrict__ fc_b,
    const bf16* __restrict__ fcw_bf, bf16* __restrict__ x_bf) {
    __shared__ __align__(16) bf16 As[128][32];
    const int tid = threadIdx.x;
    const int wave = tid >> 5, lane = tid & 31;
    const int lmod = lane & 15, lhalf = lane >> 4;
    const int m0 = blockIdx.x * 128;
    const int n0 = blockIdx.y * 64;

    v8f acc[4];
#pragma unroll
    for (int i = 0; i < 4; ++i) acc[i] = vzero8f();

    for (int kc = 0; kc < FEAT / 32; ++kc) {
        // stage A tile (f32 -> bf16) into LDS
#pragma unroll
        for (int i = 0; i < 4; ++i) {
            int fl = tid + i * 256;       // 0..1023 float4 slots
            int row = fl >> 3, c4 = fl & 7;
            const float* sp = state + (size_t)(m0 + row) * FEAT + kc * 32 + c4 * 4;
            float4 f = *(const float4*)sp;
            if (kc + 1 < FEAT / 32) __builtin_prefetch(sp + 32, 0, 1);
            bf16* dst = &As[row][c4 * 4];
            dst[0] = (bf16)f.x; dst[1] = (bf16)f.y;
            dst[2] = (bf16)f.z; dst[3] = (bf16)f.w;
        }
        __syncthreads();
        // A fragment (16-bit A 16x32 layout)
        const int am = wave * 16 + lmod;
        const int h8 = lhalf * 16;
        v8bf alo = *(const v8bf*)&As[am][h8];
        v8bf ahi = *(const v8bf*)&As[am][h8 + 8];
        v16bf afrag = mkfrag(alo, ahi);
#pragma unroll
        for (int nt = 0; nt < 4; ++nt) {
            const int n = n0 + nt * 16 + lmod;
            v16bf bfrag =
                *(const v16bf*)(fcw_bf + (size_t)n * FEAT + kc * 32 + lhalf * 16);
            acc[nt] = wmma_bf16(afrag, bfrag, acc[nt]);
        }
        __syncthreads();
    }
    // epilogue: + fc_b, write bf16 into x buffer cols [0..255]
#pragma unroll
    for (int nt = 0; nt < 4; ++nt) {
        const int n = n0 + nt * 16 + lmod;
        const float bias = fc_b[n];
#pragma unroll
        for (int v = 0; v < 8; ++v) {
            const int m = m0 + wave * 16 + v + lhalf * 8;
            x_bf[(size_t)m * XDIM + n] = (bf16)(acc[nt][v] + bias);
        }
    }
}

// ---------------- K3: input-gate GEMM  (20480 x 288) @ (288 x 1024) ----------------
__global__ __launch_bounds__(256) void xgates_kernel(
    const bf16* __restrict__ x_bf, const bf16* __restrict__ wih_bf,
    const float* __restrict__ bgate, float* __restrict__ gx) {
    const int tid = threadIdx.x;
    const int wave = tid >> 5, lane = tid & 31;
    const int lmod = lane & 15, lhalf = lane >> 4;
    const int m0 = blockIdx.x * 128;
    const int n0 = blockIdx.y * 64;

    v8f acc[4];
#pragma unroll
    for (int i = 0; i < 4; ++i) acc[i] = vzero8f();

    const bf16* arow = x_bf + (size_t)(m0 + wave * 16 + lmod) * XDIM;
#pragma unroll
    for (int kc = 0; kc < XDIM / 32; ++kc) {
        v8bf alo = *(const v8bf*)(arow + kc * 32 + lhalf * 16);
        v8bf ahi = *(const v8bf*)(arow + kc * 32 + lhalf * 16 + 8);
        v16bf afrag = mkfrag(alo, ahi);
#pragma unroll
        for (int nt = 0; nt < 4; ++nt) {
            const int n = n0 + nt * 16 + lmod;
            v16bf bfrag =
                *(const v16bf*)(wih_bf + (size_t)n * XDIM + kc * 32 + lhalf * 16);
            acc[nt] = wmma_bf16(afrag, bfrag, acc[nt]);
        }
    }
#pragma unroll
    for (int nt = 0; nt < 4; ++nt) {
        const int n = n0 + nt * 16 + lmod;
        const float bias = bgate[n];
#pragma unroll
        for (int v = 0; v < 8; ++v) {
            const int m = m0 + wave * 16 + v + lhalf * 8;
            gx[(size_t)m * GATES + n] = acc[nt][v] + bias;
        }
    }
}

// ---------------- K4: persistent LSTM scan ----------------
// 16 blocks, 16 batch rows each (batch rows are independent -> no global sync).
// wave w owns h-slice [w*32, w*32+32) and all 4 gates for it; c stays in regs
// in WMMA D-layout; h (bf16) lives in LDS as the shared A matrix.
__global__ __launch_bounds__(256) void lstm_kernel(
    const bf16* __restrict__ whh_bf, const float* __restrict__ gx,
    const unsigned char* __restrict__ done, bf16* __restrict__ hs_bf) {
    __shared__ __align__(16) bf16 h_lds[16][HID];
    const int tid = threadIdx.x;
    const int wave = tid >> 5, lane = tid & 31;
    const int lmod = lane & 15, lhalf = lane >> 4;
    const int bbase = blockIdx.x * 16;
    const int m_d = lhalf * 8;  // D-layout row base for this lane

    for (int i = tid; i < 16 * HID; i += 256) ((bf16*)h_lds)[i] = (bf16)0.0f;
    float c[2][8];
#pragma unroll
    for (int s = 0; s < 2; ++s)
#pragma unroll
        for (int v = 0; v < 8; ++v) c[s][v] = 0.0f;

    for (int t = 0; t < T_DIM; ++t) {
        __syncthreads();  // prev-step h writes visible
        // done mask: zero h rows in LDS and c in registers
        for (int i = tid; i < 16 * HID; i += 256) {
            int row = i >> 8;
            if (done[t * B_DIM + bbase + row]) ((bf16*)h_lds)[i] = (bf16)0.0f;
        }
#pragma unroll
        for (int v = 0; v < 8; ++v) {
            if (done[t * B_DIM + bbase + m_d + v]) { c[0][v] = 0.0f; c[1][v] = 0.0f; }
        }
        __syncthreads();

        v8f acc[4][2];
#pragma unroll
        for (int g = 0; g < 4; ++g)
#pragma unroll
            for (int s = 0; s < 2; ++s) acc[g][s] = vzero8f();

#pragma unroll
        for (int kc = 0; kc < HID / 32; ++kc) {
            v8bf alo = *(const v8bf*)&h_lds[lmod][kc * 32 + lhalf * 16];
            v8bf ahi = *(const v8bf*)&h_lds[lmod][kc * 32 + lhalf * 16 + 8];
            v16bf afrag = mkfrag(alo, ahi);
#pragma unroll
            for (int g = 0; g < 4; ++g) {
#pragma unroll
                for (int s = 0; s < 2; ++s) {
                    const int n = g * HID + wave * 32 + s * 16 + lmod;
                    v16bf bfrag = *(const v16bf*)(whh_bf + (size_t)n * HID +
                                                  kc * 32 + lhalf * 16);
                    acc[g][s] = wmma_bf16(afrag, bfrag, acc[g][s]);
                }
            }
        }
        __syncthreads();  // all waves done reading h before rewriting it

        const size_t rowbase = (size_t)(t * B_DIM + bbase);
#pragma unroll
        for (int s = 0; s < 2; ++s) {
            const int hn = wave * 32 + s * 16 + lmod;
#pragma unroll
            for (int v = 0; v < 8; ++v) {
                const size_t r = (rowbase + m_d + v) * GATES;
                const float gi = acc[0][s][v] + gx[r + 0 * HID + hn];
                const float gf = acc[1][s][v] + gx[r + 1 * HID + hn];
                const float gg = acc[2][s][v] + gx[r + 2 * HID + hn];
                const float go = acc[3][s][v] + gx[r + 3 * HID + hn];
                const float iv = sigmoidf_(gi);
                const float fv = sigmoidf_(gf);
                const float ov = sigmoidf_(go);
                const float gv = tanhf(gg);
                const float cn = fv * c[s][v] + iv * gv;
                c[s][v] = cn;
                const bf16 hb = (bf16)(ov * tanhf(cn));
                h_lds[m_d + v][hn] = hb;
                hs_bf[(rowbase + m_d + v) * HID + hn] = hb;
            }
        }
    }
}

// ---------------- K5: post-MLP (WMMA) + actor/critic heads (fused) ----------------
__global__ __launch_bounds__(256) void post_kernel(
    const bf16* __restrict__ hs_bf, const bf16* __restrict__ pfcw_bf,
    const float* __restrict__ pfc_b, const float* __restrict__ actor_w,
    const float* __restrict__ actor_b, const float* __restrict__ critic_w,
    const float* __restrict__ critic_b, float* __restrict__ out) {
    __shared__ __align__(16) bf16 feat[128][PHID];
    const int tid = threadIdx.x;
    const int wave = tid >> 5, lane = tid & 31;
    const int lmod = lane & 15, lhalf = lane >> 4;
    const int m0 = blockIdx.x * 128;

    v8f acc[8];
#pragma unroll
    for (int i = 0; i < 8; ++i) acc[i] = vzero8f();

    const bf16* arow = hs_bf + (size_t)(m0 + wave * 16 + lmod) * HID;
#pragma unroll
    for (int kc = 0; kc < HID / 32; ++kc) {
        v8bf alo = *(const v8bf*)(arow + kc * 32 + lhalf * 16);
        v8bf ahi = *(const v8bf*)(arow + kc * 32 + lhalf * 16 + 8);
        v16bf afrag = mkfrag(alo, ahi);
#pragma unroll
        for (int nt = 0; nt < 8; ++nt) {
            const int n = nt * 16 + lmod;
            v16bf bfrag =
                *(const v16bf*)(pfcw_bf + (size_t)n * HID + kc * 32 + lhalf * 16);
            acc[nt] = wmma_bf16(afrag, bfrag, acc[nt]);
        }
    }
#pragma unroll
    for (int nt = 0; nt < 8; ++nt) {
        const int n = nt * 16 + lmod;
        const float bias = pfc_b[n];
#pragma unroll
        for (int v = 0; v < 8; ++v) {
            float f = acc[nt][v] + bias;
            f = f > 0.0f ? f : 0.0f;  // relu
            feat[wave * 16 + v + lhalf * 8][n] = (bf16)f;
        }
    }
    __syncthreads();
    // heads: logits (18) + value (1) per row, dot over PHID=128 from LDS
    for (int idx = tid; idx < 128 * (ACT + 1); idx += 256) {
        const int row = idx / (ACT + 1);
        const int o = idx % (ACT + 1);
        const float* wrow = (o < ACT) ? (actor_w + (size_t)o * PHID) : critic_w;
        float s = (o < ACT) ? actor_b[o] : critic_b[0];
#pragma unroll 4
        for (int k = 0; k < PHID; ++k) s += (float)feat[row][k] * wrow[k];
        const int tb = m0 + row;
        if (o < ACT) out[(size_t)tb * ACT + o] = s;
        else out[(size_t)TBROWS * ACT + tb] = s;
    }
}

// ---------------- launch ----------------
extern "C" void kernel_launch(void* const* d_in, const int* in_sizes, int n_in,
                              void* d_out, int out_size, void* d_ws, size_t ws_size,
                              hipStream_t stream) {
    (void)in_sizes; (void)n_in; (void)out_size; (void)ws_size;
    const float* state  = (const float*)d_in[0];
    const float* action = (const float*)d_in[1];
    const float* reward = (const float*)d_in[2];
    const unsigned char* done = (const unsigned char*)d_in[3];
    const float* fc_w = (const float*)d_in[4];
    const float* fc_b = (const float*)d_in[5];
    const float* w_ih = (const float*)d_in[6];
    const float* w_hh = (const float*)d_in[7];
    const float* b_ih = (const float*)d_in[8];
    const float* b_hh = (const float*)d_in[9];
    const float* pfc_w = (const float*)d_in[10];
    const float* pfc_b = (const float*)d_in[11];
    const float* actor_w = (const float*)d_in[12];
    const float* actor_b = (const float*)d_in[13];
    const float* critic_w = (const float*)d_in[14];
    const float* critic_b = (const float*)d_in[15];

    bf16* wsb = (bf16*)d_ws;
    float* wsf = (float*)((char*)d_ws + BF_BYTES);
    float* out = (float*)d_out;

    prep_kernel<<<2048, 256, 0, stream>>>(fc_w, w_ih, w_hh, pfc_w, b_ih, b_hh,
                                          wsb, wsf);
    fillx_kernel<<<1024, 256, 0, stream>>>(reward, action, wsb + XBF_OFF);
    fc_gemm_kernel<<<dim3(TBROWS / 128, HID / 64), 256, 0, stream>>>(
        state, fc_b, wsb + FCW_BF_OFF, wsb + XBF_OFF);
    xgates_kernel<<<dim3(TBROWS / 128, GATES / 64), 256, 0, stream>>>(
        wsb + XBF_OFF, wsb + WIH_BF_OFF, wsf + BGATE_OFF, wsf + GX_OFF);
    lstm_kernel<<<B_DIM / 16, 256, 0, stream>>>(wsb + WHH_BF_OFF, wsf + GX_OFF,
                                                done, wsb + HSBF_OFF);
    post_kernel<<<TBROWS / 128, 256, 0, stream>>>(
        wsb + HSBF_OFF, wsb + PFCW_BF_OFF, pfc_b, actor_w, actor_b, critic_w,
        critic_b, out);
}
